// MultilinearAttention_71116068487391
// MI455X (gfx1250) — matrix operand order repaired and checked by reading, not compiled, not run on hardware
//
#include <hip/hip_runtime.h>
#include <hip/hip_bf16.h>
#include <stdint.h>

#define C_DIM 1024
#define H_DIM 16
#define D_DIM 64
#define B_DIM 4
#define N_DIM 4096
#define ROWS  (B_DIM * N_DIM)   // 16384

typedef __attribute__((ext_vector_type(16))) __bf16 v16bf;
typedef __attribute__((ext_vector_type(8)))  float  v8f;
typedef __attribute__((ext_vector_type(4)))  unsigned int v4u;
typedef __attribute__((ext_vector_type(8)))  int v8i;
typedef __attribute__((ext_vector_type(4)))  int v4i;

__device__ __forceinline__ uint16_t f2bf(float f) {
    uint32_t u = __float_as_uint(f);
    u += 0x7fffu + ((u >> 16) & 1u);
    return (uint16_t)(u >> 16);
}

union FragBF { v16bf v; uint4 q[2]; };

// A-fragment: 16x32 bf16, row-major source [M,K].
// lane<16: K {0..7} U {16..23}; lane>=16: K {8..15} U {24..31}.
__device__ __forceinline__ v16bf load_frag_a(const uint16_t* rowptr, int hi) {
    FragBF f;
    const uint16_t* p = rowptr + (hi << 3);
    f.q[0] = *reinterpret_cast<const uint4*>(p);
    f.q[1] = *reinterpret_cast<const uint4*>(p + 16);
    return f.v;
}

// B-fragment: 32x16 bf16 from transposed operand row-major [N,K].
__device__ __forceinline__ v16bf load_frag_b(const uint16_t* rowptr, int hi) {
    FragBF f;
    const uint16_t* p = rowptr + (hi << 3);
    f.q[0] = *reinterpret_cast<const uint4*>(p);
    f.q[1] = *reinterpret_cast<const uint4*>(p + 16);
    return f.v;
}

__device__ __forceinline__ v8f wmma_bf16(v16bf a, v16bf b, v8f c) {
    return __builtin_amdgcn_wmma_f32_16x16x32_bf16(false, a, false, b, (short)0, c,
                                                   false, false);
}

// ---- CDNA5 wait helpers -------------------------------------------------
#if __has_builtin(__builtin_amdgcn_s_wait_tensorcnt)
#define WAIT_TENSOR0() __builtin_amdgcn_s_wait_tensorcnt(0)
#else
#define WAIT_TENSOR0() asm volatile("s_wait_tensorcnt 0x0" ::: "memory")
#endif

#if __has_builtin(__builtin_amdgcn_s_wait_asynccnt)
#define WAIT_ASYNC(n) __builtin_amdgcn_s_wait_asynccnt(n)
#else
#define WAIT_ASYNC(n) asm volatile("s_wait_asynccnt " #n ::: "memory")
#endif

// async copy of 16 bytes per lane: global -> LDS (ASYNCcnt-tracked)
__device__ __forceinline__ void async_g2l_b128(uint32_t lds_addr, uint64_t gaddr) {
    asm volatile("global_load_async_to_lds_b128 %0, %1, off"
                 :: "v"(lds_addr), "v"(gaddr) : "memory");
}

// ---------------------------------------------------------------- converts
__global__ void convert_x_kernel(const float* __restrict__ x, uint16_t* __restrict__ xb) {
    size_t i = ((size_t)blockIdx.x * blockDim.x + threadIdx.x) * 4;
    float4 v = *reinterpret_cast<const float4*>(x + i);
    uint2 pk;
    pk.x = (uint32_t)f2bf(v.x) | ((uint32_t)f2bf(v.y) << 16);
    pk.y = (uint32_t)f2bf(v.z) | ((uint32_t)f2bf(v.w) << 16);
    *reinterpret_cast<uint2*>(xb + i) = pk;
}

// Convert + transpose six 1024x1024 fp32 weights -> bf16 Wt[d,c].
__global__ void convert_w_kernel(const float* __restrict__ Wq, const float* __restrict__ Wk,
                                 const float* __restrict__ Wv, const float* __restrict__ Wo,
                                 uint16_t* __restrict__ wT) {
    __shared__ float tile[32][33];
    const int w = blockIdx.z;
    const float* src = (w == 0) ? Wq
                     : (w <= 2) ? Wk + (size_t)(w - 1) * C_DIM * C_DIM
                     : (w <= 4) ? Wv + (size_t)(w - 3) * C_DIM * C_DIM
                                : Wo;
    const int c0 = blockIdx.y * 32;
    const int d0 = blockIdx.x * 32;
    for (int i = threadIdx.y; i < 32; i += 8)
        tile[i][threadIdx.x] = src[(size_t)(c0 + i) * C_DIM + d0 + threadIdx.x];
    __syncthreads();
    uint16_t* dst = wT + (size_t)w * C_DIM * C_DIM;
    for (int i = threadIdx.y; i < 32; i += 8)
        dst[(size_t)(d0 + i) * C_DIM + c0 + threadIdx.x] = f2bf(tile[threadIdx.x][i]);
}

__global__ void zero_states_kernel(float* __restrict__ states) {
    int i = blockIdx.x * blockDim.x + threadIdx.x;
    states[i] = 0.0f;
}

// ---------------------------------------------------------------- projections
// Block: 256 thr = 8 waves; covers 32 rows x 512 cols of one projection.
// Wave 0 DMAs the 32x1024 bf16 activation panel (64 KB) into LDS with the TDM
// (tensor_load_to_lds); every wave then computes a 32x64 output block:
// 2 A-fragments (from LDS) x 4 B-fragments (weights, L2-resident) = 8 WMMAs/k-step.
__global__ void proj_kernel(const uint16_t* __restrict__ xb, const float* __restrict__ x,
                            const uint16_t* __restrict__ wT,
                            const float* __restrict__ bq, const float* __restrict__ bk,
                            const float* __restrict__ bv,
                            uint16_t* __restrict__ qb, uint16_t* __restrict__ kvT) {
    __shared__ uint16_t apanel[32 * C_DIM];            // 64 KB
    const int lane = threadIdx.x & 31;
    const int wv   = threadIdx.x >> 5;                 // 0..7
    const int nt   = blockIdx.x;                       // 0..511
    const int cg   = blockIdx.y;                       // 0..1
    const int proj = blockIdx.z;                       // 0..4
    const int m0 = nt << 5;
    const int c0 = (cg << 9) + (wv << 6);
    const int lo = lane & 15, hi = lane >> 4;

#if __has_builtin(__builtin_amdgcn_tensor_load_to_lds)
    if (threadIdx.x < 32) {
        const uint64_t ga = (uint64_t)(uintptr_t)(xb + (size_t)m0 * C_DIM);
        const uint32_t la = (uint32_t)(uintptr_t)&apanel[0];
        // D# group0: count=1 | lds_addr | global_addr[56:0] | type=2
        v4u g0 = { 1u, la, (uint32_t)ga,
                   (uint32_t)((ga >> 32) & 0x01FFFFFFu) | (2u << 30) };
        // D# group1: data_size=2B; tensor_dim0=1024; tensor_dim1=32;
        //            tile_dim0=1024; tile_dim1=32; tensor_dim0_stride=1024
        v8i g1 = { (int)(1u << 16),
                   (int)(((uint32_t)C_DIM & 0xFFFFu) << 16),
                   (int)((((uint32_t)C_DIM >> 16) & 0xFFFFu) | (32u << 16)),
                   (int)(((uint32_t)C_DIM) << 16),
                   (int)32,
                   (int)C_DIM,
                   0, 0 };
        v4i gz4 = {0, 0, 0, 0};
        v8i gz8 = {0, 0, 0, 0, 0, 0, 0, 0};
        __builtin_amdgcn_tensor_load_to_lds(g0, g1, gz4, gz4, gz8, 0);
        WAIT_TENSOR0();
    }
#else
    for (int i = threadIdx.x; i < 32 * C_DIM / 8; i += 256)
        *reinterpret_cast<uint4*>(&apanel[i * 8]) =
            *reinterpret_cast<const uint4*>(xb + (size_t)m0 * C_DIM + i * 8);
#endif
    __syncthreads();

    const uint16_t* wbase = wT + (size_t)proj * C_DIM * C_DIM;
    const uint16_t* br[4];
#pragma unroll
    for (int j = 0; j < 4; ++j) br[j] = wbase + (size_t)(c0 + j * 16 + lo) * C_DIM;
    const uint16_t* arow0 = &apanel[lo * C_DIM];
    const uint16_t* arow1 = &apanel[(16 + lo) * C_DIM];

    v8f acc[2][4] = {};
    for (int k0 = 0; k0 < C_DIM; k0 += 32) {
        v16bf a0 = load_frag_a(arow0 + k0, hi);        // ds_load_b128 x2
        v16bf a1 = load_frag_a(arow1 + k0, hi);
#pragma unroll
        for (int j = 0; j < 4; ++j) {
            v16bf b = load_frag_b(br[j] + k0, hi);
            acc[0][j] = wmma_bf16(a0, b, acc[0][j]);
            acc[1][j] = wmma_bf16(a1, b, acc[1][j]);
        }
    }

    if (proj == 0) {
#pragma unroll
        for (int mt = 0; mt < 2; ++mt) {
            const int rbase = m0 + mt * 16 + (hi << 3);
#pragma unroll
            for (int j = 0; j < 4; ++j) {
                const int cgl = c0 + j * 16 + lo;
                const float bias = bq[cgl];
#pragma unroll
                for (int jj = 0; jj < 8; ++jj) {
                    const int row = rbase + jj;
                    const float v = acc[mt][j][jj] + x[(size_t)row * C_DIM + cgl] + bias;
                    qb[(size_t)row * C_DIM + cgl] = f2bf(v);
                }
            }
        }
    } else {
        const int p = proj - 1;                        // 0,1 = k ; 2,3 = v
        const float* bb = (p < 2) ? (bk + p * C_DIM) : (bv + (p - 2) * C_DIM);
        const float scale = 0.015625f;                 // 1/sqrt(4096)
#pragma unroll
        for (int mt = 0; mt < 2; ++mt) {
            const int rbase = m0 + mt * 16 + (hi << 3);
#pragma unroll
            for (int j = 0; j < 4; ++j) {
                const int cgl = c0 + j * 16 + lo;
                const float bias = bb[cgl];
                uint32_t w0 = 0, w1 = 0, w2 = 0, w3 = 0;
#pragma unroll
                for (int jj = 0; jj < 8; ++jj) {
                    const float v =
                        (acc[mt][j][jj] + x[(size_t)(rbase + jj) * C_DIM + cgl] + bias) * scale;
                    const uint32_t h16 = (uint32_t)f2bf(v) << ((jj & 1) * 16);
                    if (jj < 2) w0 |= h16; else if (jj < 4) w1 |= h16;
                    else if (jj < 6) w2 |= h16; else w3 |= h16;
                }
                uint4 pk = {w0, w1, w2, w3};
                *reinterpret_cast<uint4*>(kvT + ((size_t)p * C_DIM + cgl) * ROWS + rbase) = pk;
            }
        }
    }
}

// ---------------------------------------------------------------- states
// states[k,b,h,d,e] += sum_n kT[d,n] * vT[e,n]  (split-K over 8 chunks, f32 atomics)
__global__ void states_kernel(const uint16_t* __restrict__ kvT, float* __restrict__ states) {
    const int lane = threadIdx.x & 31;
    const int wid  = blockIdx.x * (blockDim.x >> 5) + (threadIdx.x >> 5);
    const int chunk = wid & 7;
    const int h = (wid >> 3) & 15;
    const int b = (wid >> 7) & 3;
    const int k = (wid >> 9) & 1;
    const int lo = lane & 15, hi = lane >> 4;

    const size_t nbase = (size_t)b * N_DIM + (size_t)chunk * 512;
    const uint16_t* abase = kvT + ((size_t)k * C_DIM + h * 64) * ROWS;
    const uint16_t* bbase = kvT + ((size_t)(2 + k) * C_DIM + h * 64) * ROWS;
    const uint16_t* ar[4];
    const uint16_t* vr[4];
#pragma unroll
    for (int t = 0; t < 4; ++t) {
        ar[t] = abase + (size_t)(t * 16 + lo) * ROWS + nbase;
        vr[t] = bbase + (size_t)(t * 16 + lo) * ROWS + nbase;
    }

    v8f acc[4][4] = {};
    for (int n0 = 0; n0 < 512; n0 += 32) {
        v16bf af[4], bf_[4];
#pragma unroll
        for (int t = 0; t < 4; ++t) {
            af[t]  = load_frag_a(ar[t] + n0, hi);
            bf_[t] = load_frag_b(vr[t] + n0, hi);
        }
#pragma unroll
        for (int mt = 0; mt < 4; ++mt)
#pragma unroll
            for (int et = 0; et < 4; ++et)
                acc[mt][et] = wmma_bf16(af[mt], bf_[et], acc[mt][et]);
    }

    float* sb = states + (((size_t)k * B_DIM + b) * H_DIM + h) * 4096;
#pragma unroll
    for (int mt = 0; mt < 4; ++mt)
#pragma unroll
        for (int et = 0; et < 4; ++et) {
            const int e = et * 16 + lo;
#pragma unroll
            for (int jj = 0; jj < 8; ++jj) {
                const int d = mt * 16 + (hi << 3) + jj;
                atomicAdd(sb + d * 64 + e, acc[mt][et][jj]);
            }
        }
}

// state = prod_k states[k]; store TRANSPOSED bf16 stT[b,h,e,d]
__global__ void stprod_kernel(const float* __restrict__ states, uint16_t* __restrict__ stT) {
    const int idx = blockIdx.x * blockDim.x + threadIdx.x;   // bh*4096 + e*64 + d
    const int d  = idx & 63;
    const int e  = (idx >> 6) & 63;
    const int bh = idx >> 12;
    const float s0 = states[(size_t)bh * 4096 + d * 64 + e];
    const float s1 = states[(size_t)(64 + bh) * 4096 + d * 64 + e];
    stT[(size_t)idx] = f2bf(s0 * s1);
}

// ---------------------------------------------------------------- outT = stateT @ qT
__global__ void outT_kernel(const uint16_t* __restrict__ stT, const uint16_t* __restrict__ qb,
                            float* __restrict__ attn) {
    const int lane = threadIdx.x & 31;
    const int wid  = blockIdx.x * (blockDim.x >> 5) + (threadIdx.x >> 5);
    const int nt = wid & 255;
    const int h  = (wid >> 8) & 15;
    const int b  = wid >> 12;
    const int lo = lane & 15, hi = lane >> 4;

    const uint16_t* abase = stT + (size_t)(b * 16 + h) * 4096;
    const uint16_t* brow  = qb + ((size_t)b * N_DIM + nt * 16 + lo) * C_DIM + h * 64;

    v8f acc[4] = {};
#pragma unroll
    for (int k0 = 0; k0 < 64; k0 += 32) {
        v16bf bf_ = load_frag_b(brow + k0, hi);
#pragma unroll
        for (int et = 0; et < 4; ++et) {
            v16bf af = load_frag_a(abase + (size_t)(et * 16 + lo) * 64 + k0, hi);
            acc[et] = wmma_bf16(af, bf_, acc[et]);
        }
    }

    const int grow = b * N_DIM + nt * 16 + lo;
    float* obase = attn + (size_t)grow * C_DIM + h * 64;
#pragma unroll
    for (int et = 0; et < 4; ++et) {
        float4 v0 = {acc[et][0], acc[et][1], acc[et][2], acc[et][3]};
        float4 v1 = {acc[et][4], acc[et][5], acc[et][6], acc[et][7]};
        float* p = obase + et * 16 + (hi << 3);
        *reinterpret_cast<float4*>(p)     = v0;
        *reinterpret_cast<float4*>(p + 4) = v1;
    }
}

// ---------------------------------------------------------------- LayerNorm -> bf16
__global__ void ln_kernel(const float* __restrict__ attn, const float* __restrict__ g,
                          const float* __restrict__ be, uint16_t* __restrict__ lnb) {
    const int row = blockIdx.x;
    const int t = threadIdx.x;
    const float4 v = *reinterpret_cast<const float4*>(attn + (size_t)row * C_DIM + t * 4);
    __shared__ float r1[256], r2[256];
    r1[t] = v.x + v.y + v.z + v.w;
    r2[t] = v.x * v.x + v.y * v.y + v.z * v.z + v.w * v.w;
    __syncthreads();
    for (int off = 128; off > 0; off >>= 1) {
        if (t < off) { r1[t] += r1[t + off]; r2[t] += r2[t + off]; }
        __syncthreads();
    }
    const float mu  = r1[0] * (1.0f / 1024.0f);
    const float var = r2[0] * (1.0f / 1024.0f) - mu * mu;
    const float rs  = rsqrtf(var + 1e-5f);
    const int c = t * 4;
    const float o0 = (v.x - mu) * rs * g[c + 0] + be[c + 0];
    const float o1 = (v.y - mu) * rs * g[c + 1] + be[c + 1];
    const float o2 = (v.z - mu) * rs * g[c + 2] + be[c + 2];
    const float o3 = (v.w - mu) * rs * g[c + 3] + be[c + 3];
    uint2 pk;
    pk.x = (uint32_t)f2bf(o0) | ((uint32_t)f2bf(o1) << 16);
    pk.y = (uint32_t)f2bf(o2) | ((uint32_t)f2bf(o3) << 16);
    *reinterpret_cast<uint2*>(lnb + (size_t)row * C_DIM + c) = pk;
}

// ---------------------------------------------------------------- final GEMM: ln @ Wo + bo
// Block: 256 thr = 8 waves covering 16 rows x 512 cols. A tiles (16x32) are
// double-buffered in LDS via async global->LDS copies issued by wave 0.
__global__ void out_gemm_kernel(const uint16_t* __restrict__ lnb, const uint16_t* __restrict__ woT,
                                const float* __restrict__ bo, float* __restrict__ out) {
    __shared__ uint16_t atile[2][16 * 32];             // 2 x 1 KB
    const int lane = threadIdx.x & 31;
    const int wv   = threadIdx.x >> 5;
    const int nt = blockIdx.x >> 1;
    const int cg = blockIdx.x & 1;
    const int m0 = nt << 4;
    const int c0 = (cg << 9) + (wv << 6);
    const int lo = lane & 15, hi = lane >> 4;

    const uint16_t* apanel_g = lnb + (size_t)m0 * C_DIM;
    const int fr  = lane >> 2;                         // fill row 0..7
    const int fc  = (lane & 3) << 3;                   // fill col 0,8,16,24

    const uint16_t* br[4];
#pragma unroll
    for (int j = 0; j < 4; ++j) br[j] = woT + (size_t)(c0 + j * 16 + lo) * C_DIM;

    v8f acc[4] = {};
    if (wv == 0) {                                      // prologue: fill buf 0
        async_g2l_b128((uint32_t)(uintptr_t)&atile[0][fr * 32 + fc],
                       (uint64_t)(uintptr_t)(apanel_g + (size_t)fr * C_DIM + fc));
        async_g2l_b128((uint32_t)(uintptr_t)&atile[0][(fr + 8) * 32 + fc],
                       (uint64_t)(uintptr_t)(apanel_g + (size_t)(fr + 8) * C_DIM + fc));
    }
    for (int k0 = 0; k0 < C_DIM; k0 += 32) {
        const int cur = (k0 >> 5) & 1;
        if (wv == 0) {
            if (k0 + 32 < C_DIM) {                      // overlap next DMA with compute
                const int kn = k0 + 32;
                async_g2l_b128((uint32_t)(uintptr_t)&atile[cur ^ 1][fr * 32 + fc],
                               (uint64_t)(uintptr_t)(apanel_g + (size_t)fr * C_DIM + kn + fc));
                async_g2l_b128((uint32_t)(uintptr_t)&atile[cur ^ 1][(fr + 8) * 32 + fc],
                               (uint64_t)(uintptr_t)(apanel_g + (size_t)(fr + 8) * C_DIM + kn + fc));
                WAIT_ASYNC(2);                          // cur buffer complete
            } else {
                WAIT_ASYNC(0);
            }
        }
        __syncthreads();
        FragBF a;
        const uint16_t* p = &atile[cur][lo * 32 + (hi << 3)];
        a.q[0] = *reinterpret_cast<const uint4*>(p);
        a.q[1] = *reinterpret_cast<const uint4*>(p + 16);
#pragma unroll
        for (int j = 0; j < 4; ++j)
            acc[j] = wmma_bf16(a.v, load_frag_b(br[j] + k0, hi), acc[j]);
        __syncthreads();                                // reads done before buf reuse
    }

    const int rbase = m0 + (hi << 3);
#pragma unroll
    for (int j = 0; j < 4; ++j) {
        const int cgl = c0 + j * 16 + lo;
        const float bias = bo[cgl];
#pragma unroll
        for (int jj = 0; jj < 8; ++jj)
            out[(size_t)(rbase + jj) * C_DIM + cgl] = acc[j][jj] + bias;
    }
}

// ---------------------------------------------------------------- launch
extern "C" void kernel_launch(void* const* d_in, const int* in_sizes, int n_in,
                              void* d_out, int out_size, void* d_ws, size_t ws_size,
                              hipStream_t stream) {
    const float* x    = (const float*)d_in[0];
    const float* Wq   = (const float*)d_in[1];
    const float* bq   = (const float*)d_in[2];
    const float* Wk   = (const float*)d_in[3];
    const float* bk   = (const float*)d_in[4];
    const float* Wv   = (const float*)d_in[5];
    const float* bv   = (const float*)d_in[6];
    const float* Wo   = (const float*)d_in[7];
    const float* bo   = (const float*)d_in[8];
    const float* ln_g = (const float*)d_in[9];
    const float* ln_b = (const float*)d_in[10];
    float* out = (float*)d_out;

    // workspace carve (~217 MB; attn & lnb alias the dead kvT region)
    char* base = (char*)d_ws;
    uint16_t* xb     = (uint16_t*)(base);                      //  32 MB
    uint16_t* qb     = (uint16_t*)(base + (size_t)33554432);   //  32 MB
    uint16_t* wT     = (uint16_t*)(base + (size_t)67108864);   //  12 MB (6 matrices)
    float*    states = (float*)   (base + (size_t)79691776);   //   2 MB
    uint16_t* stT    = (uint16_t*)(base + (size_t)81788928);   // 0.5 MB
    uint16_t* kvT    = (uint16_t*)(base + (size_t)82313216);   // 128 MB
    float*    attn   = (float*)kvT;                            //  64 MB alias (kvT dead)
    uint16_t* lnb    = (uint16_t*)(base + (size_t)82313216 + 67108864); // 32 MB alias

    convert_x_kernel<<<16384, 256, 0, stream>>>(x, xb);
    convert_w_kernel<<<dim3(32, 32, 6), dim3(32, 8), 0, stream>>>(Wq, Wk, Wv, Wo, wT);
    zero_states_kernel<<<2048, 256, 0, stream>>>(states);
    proj_kernel<<<dim3(512, 2, 5), 256, 0, stream>>>(xb, x, wT, bq, bk, bv, qb, kvT);
    states_kernel<<<128, 256, 0, stream>>>(kvT, states);
    stprod_kernel<<<1024, 256, 0, stream>>>(states, stT);
    outT_kernel<<<2048, 256, 0, stream>>>(stT, qb, attn);
    ln_kernel<<<16384, 256, 0, stream>>>(attn, ln_g, ln_b, lnb);
    out_gemm_kernel<<<2048, 256, 0, stream>>>(lnb, wT + (size_t)5 * C_DIM * C_DIM, bo, out);
}
